// TutorNet_3264175145726
// MI455X (gfx1250) — matrix operand order repaired and checked by reading, not compiled, not run on hardware
//
#include <hip/hip_runtime.h>
#include <hip/hip_bf16.h>
#include <math.h>

typedef __bf16 bf16;
typedef __attribute__((ext_vector_type(16))) __bf16 v16bf;
typedef __attribute__((ext_vector_type(8)))  float  v8f;

#define B_   16
#define SL_  512
#define D_   256
#define H_   8
#define DH_  32
#define DFF_ 1024
#define MTOT (B_*SL_)   // 8192

// ---------- helpers ----------
__device__ inline bf16 f2bf(float f) {
  unsigned u = __builtin_bit_cast(unsigned, f);
  unsigned r = (u + 0x7fffu + ((u >> 16) & 1u)) >> 16;   // RNE
  unsigned short h = (unsigned short)r;
  return __builtin_bit_cast(bf16, h);
}

// ---------- tci precompute ----------
__global__ void k_tci(const int* __restrict__ x_cat, int* __restrict__ tci, int* __restrict__ tcis) {
  int i = blockIdx.x * blockDim.x + threadIdx.x;
  if (i >= MTOT) return;
  int s = i % SL_;
  tci[i]  = x_cat[i * 5 + 1] + 1;
  tcis[i] = (s == 0) ? 0 : (x_cat[(i - 1) * 5 + 1] + 1);
}

// ---------- embedding (mode 0 = encoder/shifted, mode 1 = decoder) ----------
__global__ __launch_bounds__(256)
void k_embed(const int* __restrict__ x_cat, const float* __restrict__ x_cont,
             const int* __restrict__ x_tags, const float* __restrict__ x_tagw,
             const float* t0, const float* w0, const float* c0,
             const float* t1, const float* w1, const float* c1,
             const float* t2, const float* w2, const float* c2,
             const float* t3, const float* w3, const float* c3,
             const float* t4, const float* w4, const float* c4,
             const float* __restrict__ tag_table, const float* __restrict__ conts_W,
             const float* __restrict__ conts_b, float* __restrict__ out, int mode)
{
  __shared__ float row[64];
  __shared__ float trow[256];
  __shared__ float red[256];
  __shared__ float scale_sh;
  int tok = blockIdx.x, d = threadIdx.x;
  int s = tok % SL_;
  int src = (mode == 0) ? ((s == 0) ? -1 : tok - 1) : tok;
  const float* tbls[5] = {t0, t1, t2, t3, t4};
  const float* Ws[5]   = {w0, w1, w2, w3, w4};
  const float* bs[5]   = {c0, c1, c2, c3, c4};
  const int nfs[5]     = {64, 32, 4, 4, 4};
  float acc = 0.f;
#pragma unroll
  for (int j = 0; j < 5; ++j) {
    int idx = 0;
    if (src >= 0) {
      idx = x_cat[src * 5 + j] + 1;
      if (mode == 1 && (j == 2 || j == 3 || j == 4)) idx = 0;
    }
    int nf = nfs[j];
    if (d < nf) row[d] = tbls[j][idx * nf + d];
    __syncthreads();
    if (d == 0) {
      float ss = 0.f;
      for (int f = 0; f < nf; ++f) ss += row[f] * row[f];
      float n = sqrtf(ss);
      scale_sh = fminf(1.f, 1.f / fmaxf(n, 1e-12f));
    }
    __syncthreads();
    float t = 0.f;
    for (int f = 0; f < nf; ++f) t += row[f] * Ws[j][f * D_ + d];
    acc += t * scale_sh + bs[j][d];
    __syncthreads();
  }
  // tag term
  for (int t = 0; t < 6; ++t) {
    int ti; float w;
    if (mode == 0) {
      if (s == 0) { ti = 0; w = 0.f; }
      else { ti = x_tags[src * 6 + t] + 1; w = x_tagw[src * 6 + t]; }
    } else {
      ti = x_tags[tok * 6 + t] + 1; w = x_tagw[tok * 6 + t];
    }
    trow[d] = tag_table[ti * D_ + d];
    red[d] = trow[d] * trow[d];
    __syncthreads();
    for (int st = 128; st > 0; st >>= 1) { if (d < st) red[d] += red[d + st]; __syncthreads(); }
    float n = sqrtf(red[0]);
    float sc = fminf(1.f, 1.f / fmaxf(n, 1e-12f));
    acc += trow[d] * sc * w;
    __syncthreads();
  }
  // continuous term
  float cv[4];
#pragma unroll
  for (int f = 0; f < 4; ++f) {
    if (mode == 0) cv[f] = (s == 0) ? 0.f : x_cont[src * 4 + f];
    else           cv[f] = (f == 0 || f == 1) ? 0.f : x_cont[tok * 4 + f];
  }
  float t = conts_b[d];
#pragma unroll
  for (int f = 0; f < 4; ++f) t += cv[f] * conts_W[f * D_ + d];
  acc += t;
  out[(size_t)tok * D_ + d] = acc / 7.0f;
}

// ---------- WMMA GEMM: out = op(A[MxK] @ W[KxN] + bias [+ resid]) ----------
// 128x64 block tile, 8 waves, each wave computes a 32x32 register tile via
// 2x2 fragments -> 4 back-to-back v_wmma_f32_16x16x32_bf16 per K-step.
template<int RELU, int RESID, int STOREBF>
__global__ __launch_bounds__(256)
void k_gemm(const float* __restrict__ A, const float* __restrict__ W,
            const float* __restrict__ bias, const float* __restrict__ resid,
            float* __restrict__ outF, bf16* __restrict__ outB, int M, int N, int K)
{
  __shared__ bf16 As[128][40];
  __shared__ bf16 Bs[32][72];
  int tid  = threadIdx.x;
  int lane = tid & 31;
  int w    = tid >> 5;       // 0..7 waves
  int wm   = w & 3;          // 4 row-groups of 32 rows
  int wn   = w >> 2;         // 2 col-groups of 32 cols
  int row0 = blockIdx.x * 128;
  int col0 = blockIdx.y * 64;
  int m    = lane & 15, half = lane >> 4;
  v8f c00 = {}, c01 = {}, c10 = {}, c11 = {};
  for (int k0 = 0; k0 < K; k0 += 32) {
    // stage A (128x32) and B (32x64) as bf16; prefetch next K-slab into cache
    for (int e = tid; e < 128 * 32; e += 256) {
      int r = e >> 5, cc = e & 31;
      const float* ap = &A[(size_t)(row0 + r) * K + k0 + cc];
      As[r][cc] = f2bf(*ap);
      if (k0 + 32 < K) __builtin_prefetch(ap + 32, 0, 1);
    }
    for (int e = tid; e < 32 * 64; e += 256) {
      int r = e >> 6, cc = e & 63;
      const float* wp = &W[(size_t)(k0 + r) * N + col0 + cc];
      Bs[r][cc] = f2bf(*wp);
      if (k0 + 32 < K) __builtin_prefetch(wp + (size_t)32 * N, 0, 1);
    }
    __syncthreads();
    v16bf a0, a1, b0, b1;
#pragma unroll
    for (int v = 0; v < 8; ++v) {
      int kb = (v >> 2) * 16 + half * 8 + (v & 3) * 2;      // A 16x32 bf16 layout
      a0[2 * v]     = As[wm * 32 + m][kb];
      a0[2 * v + 1] = As[wm * 32 + m][kb + 1];
      a1[2 * v]     = As[wm * 32 + 16 + m][kb];
      a1[2 * v + 1] = As[wm * 32 + 16 + m][kb + 1];
      int k1 = (v >> 2) * 16 + half * 8 + (v & 3) * 2;      // B 32x16 bf16 layout
      b0[2 * v]     = Bs[k1][wn * 32 + m];
      b0[2 * v + 1] = Bs[k1 + 1][wn * 32 + m];
      b1[2 * v]     = Bs[k1][wn * 32 + 16 + m];
      b1[2 * v + 1] = Bs[k1 + 1][wn * 32 + 16 + m];
    }
    c00 = __builtin_amdgcn_wmma_f32_16x16x32_bf16(false, a0, false, b0, (short)0, c00, false, false);
    c01 = __builtin_amdgcn_wmma_f32_16x16x32_bf16(false, a0, false, b1, (short)0, c01, false, false);
    c10 = __builtin_amdgcn_wmma_f32_16x16x32_bf16(false, a1, false, b0, (short)0, c10, false, false);
    c11 = __builtin_amdgcn_wmma_f32_16x16x32_bf16(false, a1, false, b1, (short)0, c11, false, false);
    __syncthreads();
  }
  auto emit = [&](int rowg, int colg, float v) {
    v += bias[colg];
    if (RESID) v += resid[(size_t)rowg * N + colg];
    if (RELU)  v = fmaxf(v, 0.f);
    if (STOREBF) outB[(size_t)rowg * N + colg] = f2bf(v);
    else         outF[(size_t)rowg * N + colg] = v;
  };
#pragma unroll
  for (int r = 0; r < 8; ++r) {
    int ml   = r + 8 * half;                                 // C/D layout
    int nl   = lane & 15;
    int rowA = row0 + wm * 32 + ml;
    int rowB = rowA + 16;
    int colA = col0 + wn * 32 + nl;
    int colB = colA + 16;
    emit(rowA, colA, c00[r]);
    emit(rowA, colB, c01[r]);
    emit(rowB, colA, c10[r]);
    emit(rowB, colB, c11[r]);
  }
}

// ---------- fused attention: one wave per (b, h, 16-query tile) ----------
// mode 0: mask = (tq==tk) || (k>q) || pad     (enc self / cross with tci_s keys)
// mode 1: mask = ((tq!=tk) && (k>q)) || pad   (dec self)
__global__ __launch_bounds__(32)
void k_attn(const bf16* __restrict__ qb, const bf16* __restrict__ kb, const bf16* __restrict__ vb,
            const int* __restrict__ tciq, const int* __restrict__ tcik,
            const unsigned char* __restrict__ pad, float* __restrict__ obuf, int mode)
{
  __shared__ float S[16][SL_];
  __shared__ float rs[16];
  int bid = blockIdx.x;
  int qt  = bid & 31;          // SL/16 = 32
  int h   = (bid >> 5) & 7;
  int b   = bid >> 8;
  int lane = threadIdx.x;
  int m = lane & 15, half = lane >> 4;
  int q0 = qt * 16;
  v16bf qf;
#pragma unroll
  for (int v = 0; v < 8; ++v) {
    int kk = (v >> 2) * 16 + half * 8 + (v & 3) * 2;
    size_t ad = ((size_t)(b * SL_ + q0 + m) * D_) + h * DH_ + kk;
    qf[2 * v] = qb[ad]; qf[2 * v + 1] = qb[ad + 1];
  }
  int tq[8];
#pragma unroll
  for (int r = 0; r < 8; ++r) tq[r] = tciq[b * SL_ + q0 + r + 8 * half];
  const float scale = 0.1767766952966369f;  // 1/sqrt(32)
  for (int kt = 0; kt < SL_ / 16; ++kt) {
    int key_n = kt * 16 + m;     // this lane's key column (B-fragment n)
    v16bf kf;
#pragma unroll
    for (int v = 0; v < 8; ++v) {
      int kd = (v >> 2) * 16 + half * 8 + (v & 3) * 2;
      size_t ad = ((size_t)(b * SL_ + key_n) * D_) + h * DH_ + kd;  // K^T: B(kdim,key)=K[key][kdim]
      kf[2 * v] = kb[ad]; kf[2 * v + 1] = kb[ad + 1];
    }
    v8f sc = {};
    sc = __builtin_amdgcn_wmma_f32_16x16x32_bf16(false, qf, false, kf, (short)0, sc, false, false);
    int tk = tcik[b * SL_ + key_n];
    int pd = pad[b * SL_ + key_n];
#pragma unroll
    for (int r = 0; r < 8; ++r) {
      int ml = r + 8 * half;
      int qg = q0 + ml;
      int kg = kt * 16 + (lane & 15);
      bool mk;
      if (mode == 0) mk = (tq[r] == tk) || (kg > qg) || (pd != 0);
      else           mk = ((tq[r] != tk) && (kg > qg)) || (pd != 0);
      S[ml][kg] = mk ? -1e9f : sc[r] * scale;
    }
  }
  __syncthreads();
  if (lane < 16) {
    float mx = -3.4e38f;
    for (int k = 0; k < SL_; ++k) mx = fmaxf(mx, S[lane][k]);
    float sum = 0.f;
    for (int k = 0; k < SL_; ++k) { float e = __expf(S[lane][k] - mx); S[lane][k] = e; sum += e; }
    rs[lane] = sum;
  }
  __syncthreads();
  v8f o0 = {}, o1 = {};
  for (int kk0 = 0; kk0 < SL_; kk0 += 32) {
    v16bf pf;
#pragma unroll
    for (int v = 0; v < 8; ++v) {
      int kk = (v >> 2) * 16 + half * 8 + (v & 3) * 2;
      pf[2 * v]     = f2bf(S[m][kk0 + kk]);
      pf[2 * v + 1] = f2bf(S[m][kk0 + kk + 1]);
    }
    v16bf v0f, v1f;
#pragma unroll
    for (int v = 0; v < 8; ++v) {
      int kd = kk0 + (v >> 2) * 16 + half * 8 + (v & 3) * 2;
      size_t a0 = ((size_t)(b * SL_ + kd) * D_)     + h * DH_ + (lane & 15);
      size_t a1 = ((size_t)(b * SL_ + kd + 1) * D_) + h * DH_ + (lane & 15);
      v0f[2 * v] = vb[a0];      v0f[2 * v + 1] = vb[a1];
      v1f[2 * v] = vb[a0 + 16]; v1f[2 * v + 1] = vb[a1 + 16];
    }
    o0 = __builtin_amdgcn_wmma_f32_16x16x32_bf16(false, pf, false, v0f, (short)0, o0, false, false);
    o1 = __builtin_amdgcn_wmma_f32_16x16x32_bf16(false, pf, false, v1f, (short)0, o1, false, false);
  }
#pragma unroll
  for (int r = 0; r < 8; ++r) {
    int ml = r + 8 * half;
    float inv = 1.0f / rs[ml];
    size_t ad = ((size_t)(b * SL_ + q0 + ml) * D_) + h * DH_ + (lane & 15);
    obuf[ad]      = o0[r] * inv;
    obuf[ad + 16] = o1[r] * inv;
  }
}

// ---------- layernorm ----------
__global__ __launch_bounds__(256)
void k_ln(const float* __restrict__ x, const float* __restrict__ w,
          const float* __restrict__ bb, float* __restrict__ out)
{
  __shared__ float red[256];
  int tok = blockIdx.x, d = threadIdx.x;
  float v = x[(size_t)tok * D_ + d];
  red[d] = v; __syncthreads();
  for (int s = 128; s > 0; s >>= 1) { if (d < s) red[d] += red[d + s]; __syncthreads(); }
  float mean = red[0] / D_;
  __syncthreads();
  float dv = v - mean;
  red[d] = dv * dv; __syncthreads();
  for (int s = 128; s > 0; s >>= 1) { if (d < s) red[d] += red[d + s]; __syncthreads(); }
  float var = red[0] / D_;
  out[(size_t)tok * D_ + d] = dv * rsqrtf(var + 1e-5f) * w[d] + bb[d];
}

// ---------- final head D -> 6 ----------
__global__ void k_head(const float* __restrict__ y, const float* __restrict__ W,
                       const float* __restrict__ bb, float* __restrict__ out)
{
  int i = blockIdx.x * blockDim.x + threadIdx.x;
  if (i >= MTOT * 6) return;
  int tok = i / 6, o = i % 6;
  float acc = bb[o];
  for (int d = 0; d < D_; ++d) acc += y[(size_t)tok * D_ + d] * W[d * 6 + o];
  out[i] = acc;
}

// ---------- host side ----------
static inline void gemm_bf(const float* A, const float* W, const float* bias, bf16* out,
                           int M, int N, int K, hipStream_t s) {
  dim3 g(M / 128, N / 64);
  k_gemm<0, 0, 1><<<g, 256, 0, s>>>(A, W, bias, nullptr, nullptr, out, M, N, K);
}
static inline void gemm_res(const float* A, const float* W, const float* bias, const float* resid,
                            float* out, int M, int N, int K, hipStream_t s) {
  dim3 g(M / 128, N / 64);
  k_gemm<0, 1, 0><<<g, 256, 0, s>>>(A, W, bias, resid, out, nullptr, M, N, K);
}
static inline void gemm_relu(const float* A, const float* W, const float* bias, float* out,
                             int M, int N, int K, hipStream_t s) {
  dim3 g(M / 128, N / 64);
  k_gemm<1, 0, 0><<<g, 256, 0, s>>>(A, W, bias, nullptr, out, nullptr, M, N, K);
}

extern "C" void kernel_launch(void* const* d_in, const int* in_sizes, int n_in,
                              void* d_out, int out_size, void* d_ws, size_t ws_size,
                              hipStream_t stream) {
  (void)in_sizes; (void)n_in; (void)out_size; (void)ws_size;
  const unsigned char* x_mask = (const unsigned char*)d_in[0];
  const int*   x_cat  = (const int*)d_in[1];
  const float* x_cont = (const float*)d_in[2];
  const int*   x_tags = (const int*)d_in[3];
  const float* x_tagw = (const float*)d_in[4];
  const float* etbl[5], *eW[5], *eb[5];
  for (int j = 0; j < 5; ++j) {
    etbl[j] = (const float*)d_in[5 + 3 * j];
    eW[j]   = (const float*)d_in[6 + 3 * j];
    eb[j]   = (const float*)d_in[7 + 3 * j];
  }
  const float* tag_table = (const float*)d_in[20];
  const float* conts_W   = (const float*)d_in[21];
  const float* conts_b   = (const float*)d_in[22];
  // encoder layer stack (insertion order)
  const float* eWq = (const float*)d_in[23], *eWk = (const float*)d_in[24];
  const float* eWv = (const float*)d_in[25], *eWo = (const float*)d_in[26];
  const float* ebq = (const float*)d_in[27], *ebk = (const float*)d_in[28];
  const float* ebv = (const float*)d_in[29], *ebo = (const float*)d_in[30];
  const float* eW1 = (const float*)d_in[31], *eb1 = (const float*)d_in[32];
  const float* eW2 = (const float*)d_in[33], *eb2 = (const float*)d_in[34];
  const float* eln0w = (const float*)d_in[35], *eln0b = (const float*)d_in[36];
  const float* eln1w = (const float*)d_in[37], *eln1b = (const float*)d_in[38];
  // decoder layer stack
  const float* dWq = (const float*)d_in[39], *dWk = (const float*)d_in[40];
  const float* dWv = (const float*)d_in[41], *dWo = (const float*)d_in[42];
  const float* dbq = (const float*)d_in[43], *dbk = (const float*)d_in[44];
  const float* dbv = (const float*)d_in[45], *dbo = (const float*)d_in[46];
  const float* cWq = (const float*)d_in[47], *cWk = (const float*)d_in[48];
  const float* cWv = (const float*)d_in[49], *cWo = (const float*)d_in[50];
  const float* cbq = (const float*)d_in[51], *cbk = (const float*)d_in[52];
  const float* cbv = (const float*)d_in[53], *cbo = (const float*)d_in[54];
  const float* dW1 = (const float*)d_in[55], *db1 = (const float*)d_in[56];
  const float* dW2 = (const float*)d_in[57], *db2 = (const float*)d_in[58];
  const float* dln0w = (const float*)d_in[59], *dln0b = (const float*)d_in[60];
  const float* dln1w = (const float*)d_in[61], *dln1b = (const float*)d_in[62];
  const float* dln2w = (const float*)d_in[63], *dln2b = (const float*)d_in[64];
  const float* enc_nw = (const float*)d_in[65], *enc_nb = (const float*)d_in[66];
  const float* dec_nw = (const float*)d_in[67], *dec_nb = (const float*)d_in[68];
  const float* mlpW = (const float*)d_in[69], *mlpb = (const float*)d_in[70];

  // workspace carve
  char* ws = (char*)d_ws;
  size_t off = 0;
  auto carve = [&](size_t bytes) { char* p = ws + off; off += (bytes + 255) & ~(size_t)255; return p; };
  int*   tci  = (int*)carve((size_t)MTOT * 4);
  int*   tcis = (int*)carve((size_t)MTOT * 4);
  float* xb   = (float*)carve((size_t)MTOT * D_ * 4);
  float* yb   = (float*)carve((size_t)MTOT * D_ * 4);
  float* memb = (float*)carve((size_t)MTOT * D_ * 4);
  float* tmp  = (float*)carve((size_t)MTOT * D_ * 4);
  bf16*  qbw  = (bf16*)carve((size_t)MTOT * D_ * 2);
  bf16*  kbw  = (bf16*)carve((size_t)MTOT * D_ * 2);
  bf16*  vbw  = (bf16*)carve((size_t)MTOT * D_ * 2);
  float* ob   = (float*)carve((size_t)MTOT * D_ * 4);
  float* ff   = (float*)carve((size_t)MTOT * DFF_ * 4);

  const int M = MTOT, NATT = B_ * H_ * (SL_ / 16);

  k_tci<<<(MTOT + 255) / 256, 256, 0, stream>>>(x_cat, tci, tcis);
  k_embed<<<MTOT, 256, 0, stream>>>(x_cat, x_cont, x_tags, x_tagw,
      etbl[0], eW[0], eb[0], etbl[1], eW[1], eb[1], etbl[2], eW[2], eb[2],
      etbl[3], eW[3], eb[3], etbl[4], eW[4], eb[4],
      tag_table, conts_W, conts_b, xb, 0);
  k_embed<<<MTOT, 256, 0, stream>>>(x_cat, x_cont, x_tags, x_tagw,
      etbl[0], eW[0], eb[0], etbl[1], eW[1], eb[1], etbl[2], eW[2], eb[2],
      etbl[3], eW[3], eb[3], etbl[4], eW[4], eb[4],
      tag_table, conts_W, conts_b, yb, 1);

  // encoder
  for (int i = 0; i < 2; ++i) {
    size_t wo = (size_t)i * D_ * D_, bo = (size_t)i * D_;
    size_t w1o = (size_t)i * D_ * DFF_, b1o = (size_t)i * DFF_;
    gemm_bf(xb, eWq + wo, ebq + bo, qbw, M, D_, D_, stream);
    gemm_bf(xb, eWk + wo, ebk + bo, kbw, M, D_, D_, stream);
    gemm_bf(xb, eWv + wo, ebv + bo, vbw, M, D_, D_, stream);
    k_attn<<<NATT, 32, 0, stream>>>(qbw, kbw, vbw, tci, tcis, x_mask, ob, 0);
    gemm_res(ob, eWo + wo, ebo + bo, xb, tmp, M, D_, D_, stream);
    k_ln<<<MTOT, 256, 0, stream>>>(tmp, eln0w + bo, eln0b + bo, xb);
    gemm_relu(xb, eW1 + w1o, eb1 + b1o, ff, M, DFF_, D_, stream);
    gemm_res(ff, eW2 + w1o, eb2 + bo, xb, tmp, M, D_, DFF_, stream);
    k_ln<<<MTOT, 256, 0, stream>>>(tmp, eln1w + bo, eln1b + bo, xb);
  }
  k_ln<<<MTOT, 256, 0, stream>>>(xb, enc_nw, enc_nb, memb);

  // decoder
  for (int i = 0; i < 2; ++i) {
    size_t wo = (size_t)i * D_ * D_, bo = (size_t)i * D_;
    size_t w1o = (size_t)i * D_ * DFF_, b1o = (size_t)i * DFF_;
    // self attention (dec mask, tci both sides)
    gemm_bf(yb, dWq + wo, dbq + bo, qbw, M, D_, D_, stream);
    gemm_bf(yb, dWk + wo, dbk + bo, kbw, M, D_, D_, stream);
    gemm_bf(yb, dWv + wo, dbv + bo, vbw, M, D_, D_, stream);
    k_attn<<<NATT, 32, 0, stream>>>(qbw, kbw, vbw, tci, tci, x_mask, ob, 1);
    gemm_res(ob, dWo + wo, dbo + bo, yb, tmp, M, D_, D_, stream);
    k_ln<<<MTOT, 256, 0, stream>>>(tmp, dln0w + bo, dln0b + bo, yb);
    // cross attention (mem mask: query tci vs key tci_s)
    gemm_bf(yb,   cWq + wo, cbq + bo, qbw, M, D_, D_, stream);
    gemm_bf(memb, cWk + wo, cbk + bo, kbw, M, D_, D_, stream);
    gemm_bf(memb, cWv + wo, cbv + bo, vbw, M, D_, D_, stream);
    k_attn<<<NATT, 32, 0, stream>>>(qbw, kbw, vbw, tci, tcis, x_mask, ob, 0);
    gemm_res(ob, cWo + wo, cbo + bo, yb, tmp, M, D_, D_, stream);
    k_ln<<<MTOT, 256, 0, stream>>>(tmp, dln1w + bo, dln1b + bo, yb);
    // ffn
    gemm_relu(yb, dW1 + w1o, db1 + b1o, ff, M, DFF_, D_, stream);
    gemm_res(ff, dW2 + w1o, db2 + bo, yb, tmp, M, D_, DFF_, stream);
    k_ln<<<MTOT, 256, 0, stream>>>(tmp, dln2w + bo, dln2b + bo, yb);
  }
  k_ln<<<MTOT, 256, 0, stream>>>(yb, dec_nw, dec_nb, tmp);
  k_head<<<(MTOT * 6 + 255) / 256, 256, 0, stream>>>(tmp, mlpW, mlpb, (float*)d_out);
}
